// NeuroBLASTAttention_32040456029011
// MI455X (gfx1250) — matrix operand order repaired and checked by reading, not compiled, not run on hardware
//
#include <hip/hip_runtime.h>
#include <hip/hip_bf16.h>

// ---------------------------------------------------------------------------
// MI455X (gfx1250) attention block: QKV proj + RMSNorm + RoPE + causal GQA
// flash attention + output proj. All GEMMs via v_wmma_f32_16x16x32_f16
// (f16 operands, fp32 accumulate). wave32 throughout. K/V and f16-A tile
// staging uses GLOBAL_LOAD_ASYNC_TO_LDS_B128 (ASYNCcnt) when available.
// ---------------------------------------------------------------------------

typedef __attribute__((ext_vector_type(16))) _Float16 v16h;
typedef __attribute__((ext_vector_type(8)))  _Float16 v8h;
typedef __attribute__((ext_vector_type(2)))  _Float16 v2h;
typedef __attribute__((ext_vector_type(8)))  float    v8f;
typedef __attribute__((ext_vector_type(4)))  float    v4f;
typedef __attribute__((ext_vector_type(4)))  int      v4i;

union V16 { v16h v; v8h h[2]; };

#define B_   2
#define S_   2048
#define HID_ 2048
#define H_   16
#define KVH_ 4
#define D_   128

#if __has_builtin(__builtin_amdgcn_global_load_async_to_lds_b128) && \
    __has_builtin(__builtin_amdgcn_s_wait_asynccnt)
#define ASYNC_LDS 1
#else
#define ASYNC_LDS 0
#endif

__device__ __forceinline__ void async_copy_b128(const void* g, void* l) {
#if ASYNC_LDS
  // signature (probe-derived): (global v4i*, lds v4i*, imm offset, imm cpol)
  __builtin_amdgcn_global_load_async_to_lds_b128(
      (__attribute__((address_space(1))) v4i*)(unsigned long long)g,
      (__attribute__((address_space(3))) v4i*)l, 0, 0);
#else
  (void)g; (void)l;
#endif
}
__device__ __forceinline__ void async_wait0() {
#if ASYNC_LDS
  __builtin_amdgcn_s_wait_asynccnt(0);
#endif
}

__device__ __forceinline__ v8f wmma_f16(v16h a, v16h b, v8f c) {
  // D = A(16x32 f16) * B(32x16 f16) + C(16x16 f32)
  return __builtin_amdgcn_wmma_f32_16x16x32_f16(
      /*neg_a=*/false, a, /*neg_b=*/false, b,
      /*c_mod=*/(short)0, c, /*reuse_a=*/false, /*reuse_b=*/false);
}

enum { EPI_OUT = 0, EPI_QK = 1, EPI_V = 2 };

// ---------------------------------------------------------------------------
// Tiled WMMA GEMM: C(MxN) = A(MxK) * B(KxN), block tile 128x128, K-step 32.
// 256 threads = 8 waves in a 2(M) x 4(N) grid; each wave computes 64x32 =
// 4x2 WMMA tiles. A staged to LDS row-major f16 (async copy when A already
// f16), B staged transposed [n][k] with on-the-fly f32->f16 conversion.
// Double-buffered LDS, next-tile staging overlapped with current compute.
// ---------------------------------------------------------------------------
template <bool AHALF, int EPI>
__global__ __launch_bounds__(256)
void gemm_kernel(const void* __restrict__ Aptr, const float* __restrict__ Bw,
                 void* __restrict__ Cout, int M, int N, int K,
                 const float* __restrict__ norm_w,
                 const float* __restrict__ cosT, const float* __restrict__ sinT,
                 int heads) {
  constexpr int BM = 128, BN = 128, BK = 32;
  constexpr int AP = 40;   // LDS pitch in halves (32 + pad, 16B aligned rows)
  constexpr int BP = 40;

  __shared__ __attribute__((aligned(16))) unsigned char smem[2 * BM * AP * 2 + 2 * BN * BP * 2];
  _Float16* As = (_Float16*)smem;                // [2][128][40]
  _Float16* Bs = As + 2 * BM * AP;               // [2][128][40] stored [n][k]

  const int tid  = threadIdx.x;
  const int lane = tid & 31;
  const int w    = tid >> 5;
  const int lh   = lane & 15;
  const int l16  = lane >> 4;
  const int wm   = w & 1;        // 0..1  (M half)
  const int wn   = w >> 1;       // 0..3  (N quarter)
  const int m0   = blockIdx.y * BM;
  const int n0   = blockIdx.x * BN;

  // ---- staging index setup ----
  const int ar = tid >> 1;            // A: row 0..127
  const int ak = (tid & 1) * 16;      // A: k-half 0/16
  const int bkp = tid >> 4;           // B: k-pair 0..15 (k = 2*bkp)
  const int bn  = (tid & 15) * 8;     // B: col chunk

  v4f areg[4]; v8h aregh[2]; v4f breg[4];

  auto loadA = [&](int kt, int buf) {
    (void)buf;
    const int k0 = kt * BK;
    if constexpr (AHALF) {
      const _Float16* Ag = (const _Float16*)Aptr + (size_t)(m0 + ar) * K + k0 + ak;
#if ASYNC_LDS
      _Float16* dst = As + buf * (BM * AP) + ar * AP + ak;
      async_copy_b128(Ag, dst);
      async_copy_b128(Ag + 8, dst + 8);
#else
      aregh[0] = *(const v8h*)(Ag);
      aregh[1] = *(const v8h*)(Ag + 8);
#endif
    } else {
      const float* Ag = (const float*)Aptr + (size_t)(m0 + ar) * K + k0 + ak;
      areg[0] = *(const v4f*)(Ag);
      areg[1] = *(const v4f*)(Ag + 4);
      areg[2] = *(const v4f*)(Ag + 8);
      areg[3] = *(const v4f*)(Ag + 12);
    }
  };
  auto storeA = [&](int buf) {
    _Float16* dst = As + buf * (BM * AP) + ar * AP + ak;
    if constexpr (AHALF) {
#if !ASYNC_LDS
      *(v8h*)dst = aregh[0]; *(v8h*)(dst + 8) = aregh[1];
#else
      (void)dst;
#endif
    } else {
      v8h h0, h1;
      #pragma unroll
      for (int i = 0; i < 4; ++i) {
        h0[i]     = (_Float16)areg[0][i];
        h0[i + 4] = (_Float16)areg[1][i];
        h1[i]     = (_Float16)areg[2][i];
        h1[i + 4] = (_Float16)areg[3][i];
      }
      *(v8h*)dst = h0; *(v8h*)(dst + 8) = h1;
    }
  };
  auto loadB = [&](int kt) {
    const float* Bg = Bw + (size_t)(kt * BK + 2 * bkp) * N + n0 + bn;
    breg[0] = *(const v4f*)(Bg);
    breg[1] = *(const v4f*)(Bg + 4);
    breg[2] = *(const v4f*)(Bg + N);
    breg[3] = *(const v4f*)(Bg + N + 4);
  };
  auto storeB = [&](int buf) {
    _Float16* dst = Bs + buf * (BN * BP) + 2 * bkp;
    #pragma unroll
    for (int i = 0; i < 8; ++i) {
      float r0 = (i < 4) ? breg[0][i] : breg[1][i - 4];
      float r1 = (i < 4) ? breg[2][i] : breg[3][i - 4];
      v2h p; p[0] = (_Float16)r0; p[1] = (_Float16)r1;
      *(v2h*)(dst + (bn + i) * BP) = p;   // Bs[n][2*bkp .. +1]
    }
  };

  v8f acc[4][2];
  const v8f vzero = {0, 0, 0, 0, 0, 0, 0, 0};
  #pragma unroll
  for (int ms = 0; ms < 4; ++ms)
    #pragma unroll
    for (int ns = 0; ns < 2; ++ns) acc[ms][ns] = vzero;

  auto compute = [&](int buf) {
    const _Float16* Ab = As + buf * (BM * AP);
    const _Float16* Bb = Bs + buf * (BN * BP);
    v16h af[4], bf[2];
    const int c0 = l16 ? 8 : 0;    // A-frag: lane<16 -> K{0-7,16-23}; else {8-15,24-31}
    #pragma unroll
    for (int ms = 0; ms < 4; ++ms) {
      const _Float16* p = Ab + (wm * 64 + ms * 16 + lh) * AP;
      V16 t; t.h[0] = *(const v8h*)(p + c0); t.h[1] = *(const v8h*)(p + c0 + 16);
      af[ms] = t.v;
    }
    const int kc = l16 ? 8 : 0;    // B-frag: lane<16 -> K{0-7,16-23}; else {8-15,24-31}
    #pragma unroll
    for (int ns = 0; ns < 2; ++ns) {
      const _Float16* p = Bb + (wn * 32 + ns * 16 + lh) * BP + kc;
      V16 t; t.h[0] = *(const v8h*)(p); t.h[1] = *(const v8h*)(p + 16);
      bf[ns] = t.v;
    }
    #pragma unroll
    for (int ms = 0; ms < 4; ++ms)
      #pragma unroll
      for (int ns = 0; ns < 2; ++ns)
        acc[ms][ns] = wmma_f16(af[ms], bf[ns], acc[ms][ns]);
  };

  const int nk = K / BK;
  loadA(0, 0); loadB(0); storeA(0); storeB(0);
  if constexpr (AHALF) async_wait0();
  __syncthreads();
  for (int kt = 0; kt < nk; ++kt) {
    const int cur = kt & 1;
    const bool more = (kt + 1) < nk;
    if (more) { loadA(kt + 1, 1 - cur); loadB(kt + 1); }   // hide global latency
    compute(cur);
    if (more) { storeA(1 - cur); storeB(1 - cur); }
    if constexpr (AHALF) async_wait0();
    __syncthreads();
  }

  // ------------------------------ epilogues ------------------------------
  if constexpr (EPI == EPI_OUT) {
    float* Co = (float*)Cout;
    #pragma unroll
    for (int ms = 0; ms < 4; ++ms)
      #pragma unroll
      for (int ns = 0; ns < 2; ++ns)
        #pragma unroll
        for (int v = 0; v < 8; ++v) {
          int r = m0 + wm * 64 + ms * 16 + v + 8 * l16;
          int c = n0 + wn * 32 + ns * 16 + lh;
          Co[(size_t)r * N + c] = acc[ms][ns][v];
        }
  } else if constexpr (EPI == EPI_V) {
    // V stored TRANSPOSED: (B, KVH, D, S) f16 so attention staging is a
    // straight (async-able) copy. The 8 accum rows per fragment are 8
    // consecutive tokens -> one aligned b128 store per fragment column.
    _Float16* Vo = (_Float16*)Cout;
    const int head = n0 >> 7;       // N-tile (=128) spans exactly one head
    #pragma unroll
    for (int ms = 0; ms < 4; ++ms) {
      int token0 = m0 + wm * 64 + ms * 16 + 8 * l16;   // v = 0 row
      int bb = token0 >> 11, ss0 = token0 & (S_ - 1);
      #pragma unroll
      for (int ns = 0; ns < 2; ++ns) {
        int d = wn * 32 + ns * 16 + lh;
        v8h pk;
        #pragma unroll
        for (int v = 0; v < 8; ++v) pk[v] = (_Float16)acc[ms][ns][v];
        *(v8h*)&Vo[(((size_t)bb * heads + head) * D_ + d) * S_ + ss0] = pk;
      }
    }
  } else {  // EPI_QK: RMSNorm over the head dim (=N tile) + RoPE -> f16
    float* red = (float*)smem;                     // [128][4] partial sumsq
    _Float16* tile = (_Float16*)(smem + 2048);     // [128][136] normalized f16
    #pragma unroll
    for (int ms = 0; ms < 4; ++ms)
      #pragma unroll
      for (int v = 0; v < 8; ++v) {
        float sq = acc[ms][0][v] * acc[ms][0][v] + acc[ms][1][v] * acc[ms][1][v];
        sq += __shfl_xor(sq, 1, 16);
        sq += __shfl_xor(sq, 2, 16);
        sq += __shfl_xor(sq, 4, 16);
        sq += __shfl_xor(sq, 8, 16);
        if (lh == 0) red[(wm * 64 + ms * 16 + v + 8 * l16) * 4 + wn] = sq;
      }
    __syncthreads();
    const float w0 = norm_w[wn * 32 + lh];
    const float w1 = norm_w[wn * 32 + 16 + lh];
    #pragma unroll
    for (int ms = 0; ms < 4; ++ms)
      #pragma unroll
      for (int v = 0; v < 8; ++v) {
        int rt = wm * 64 + ms * 16 + v + 8 * l16;
        v4f rv = *(const v4f*)&red[rt * 4];
        float var = (rv[0] + rv[1] + rv[2] + rv[3]) * (1.0f / 128.0f);
        float rstd = rsqrtf(var + 1e-6f);
        tile[rt * 136 + wn * 32 + lh]      = (_Float16)(acc[ms][0][v] * rstd * w0);
        tile[rt * 136 + wn * 32 + 16 + lh] = (_Float16)(acc[ms][1][v] * rstd * w1);
      }
    __syncthreads();
    // RoPE: out[d] = x[d]*cos[d] + (d<64 ? -x[d+64] : x[d-64]) * sin[d]
    const int r = tid >> 1;
    const int dbase = (tid & 1) * 64;
    const int token = m0 + r;
    const int bb = token >> 11, ss = token & (S_ - 1);
    const int head = n0 >> 7;
    const float* cp = cosT + (size_t)token * D_;
    const float* sp = sinT + (size_t)token * D_;
    _Float16* outp = (_Float16*)Cout + (((size_t)bb * heads + head) * S_ + ss) * D_;
    #pragma unroll 8
    for (int i = 0; i < 64; ++i) {
      int d = dbase + i;
      float x  = (float)tile[r * 136 + d];
      float pr = (float)tile[r * 136 + (d ^ 64)];
      float sg = (d < 64) ? -1.0f : 1.0f;
      outp[d] = (_Float16)(x * cp[d] + sg * pr * sp[d]);
    }
  }
}

// ---------------------------------------------------------------------------
// Flash attention: one block per (128-row q-tile, b*h). 8 waves, each owns 16
// q rows; Q A-fragments live in registers; 32-key K/V tiles stream through
// double-buffered LDS (async copy when available; V is pre-transposed so
// both K and V staging are straight b128 copies); online softmax in fp32;
// P repacked to A-layout via wave-private LDS. 16 WMMAs per key tile.
// ---------------------------------------------------------------------------
__global__ __launch_bounds__(256)
void attn_kernel(const _Float16* __restrict__ Qh, const _Float16* __restrict__ Kh,
                 const _Float16* __restrict__ Vt, _Float16* __restrict__ Ah) {
  __shared__ _Float16 Ks[2][32][136];   // [key][kd] row-major
  __shared__ _Float16 Vs[2][128][40];   // [d][key]
  __shared__ _Float16 Ps[8][16][40];    // per-wave P scratch [qrow][key]

  const int qt = blockIdx.x;           // 0..15
  const int bh = blockIdx.y;           // 0..31
  const int b  = bh >> 4, h = bh & 15;
  const int kvh = h >> 2;              // GQA: 4 q heads per kv head
  const int tid = threadIdx.x, lane = tid & 31, w = tid >> 5;
  const int lh = lane & 15, l16 = lane >> 4;
  const int qrow0 = qt * 128 + w * 16;

  const _Float16* Qbase = Qh + ((size_t)b * H_ + h) * S_ * D_;
  const _Float16* Kbase = Kh + ((size_t)b * KVH_ + kvh) * S_ * D_;
  const _Float16* Vbase = Vt + ((size_t)b * KVH_ + kvh) * D_ * S_;  // [d][s]

  // Q A-fragments (16 rows x 128 kd = 4 frags of K=32), resident in VGPRs
  v16h qf[4];
  #pragma unroll
  for (int c = 0; c < 4; ++c) {
    const _Float16* qp = Qbase + (size_t)(qrow0 + lh) * D_ + c * 32;
    V16 t;
    t.h[0] = *(const v8h*)(qp + (l16 ? 8 : 0));
    t.h[1] = *(const v8h*)(qp + 16 + (l16 ? 8 : 0));
    qf[c] = t.v;
  }

  const v8f vzero = {0, 0, 0, 0, 0, 0, 0, 0};
  v8f o[8];
  #pragma unroll
  for (int n = 0; n < 8; ++n) o[n] = vzero;
  float m8[8], l8[8], corr8[8];
  #pragma unroll
  for (int v = 0; v < 8; ++v) { m8[v] = -1e30f; l8[v] = 0.0f; }

  // staging indices: K: 32 rows x 128 kd; V: 128 rows (d) x 32 keys
  const int skey = tid >> 3, skc = (tid & 7) * 16;
  const int vd = tid >> 1,  vch = (tid & 1) * 16;
  v8h kreg0, kreg1, vreg0, vreg1;

  auto stageAsync = [&](int j, int buf) {
    const _Float16* kp = Kbase + (size_t)(j * 32 + skey) * D_ + skc;
    async_copy_b128(kp,     &Ks[buf][skey][skc]);
    async_copy_b128(kp + 8, &Ks[buf][skey][skc + 8]);
    const _Float16* vp = Vbase + (size_t)vd * S_ + j * 32 + vch;
    async_copy_b128(vp,     &Vs[buf][vd][vch]);
    async_copy_b128(vp + 8, &Vs[buf][vd][vch + 8]);
  };
  auto loadKV = [&](int j) {
    const _Float16* kp = Kbase + (size_t)(j * 32 + skey) * D_ + skc;
    kreg0 = *(const v8h*)(kp); kreg1 = *(const v8h*)(kp + 8);
    const _Float16* vp = Vbase + (size_t)vd * S_ + j * 32 + vch;
    vreg0 = *(const v8h*)(vp); vreg1 = *(const v8h*)(vp + 8);
  };
  auto storeKV = [&](int buf) {
    *(v8h*)&Ks[buf][skey][skc]     = kreg0;
    *(v8h*)&Ks[buf][skey][skc + 8] = kreg1;
    *(v8h*)&Vs[buf][vd][vch]       = vreg0;
    *(v8h*)&Vs[buf][vd][vch + 8]   = vreg1;
  };

  const float scale = 0.08838834764831845f;   // 1/sqrt(128)
  const float LOG2E = 1.4426950408889634f;

  const int nkt = qt * 4 + 4;                 // causal: keys <= qtile end
#if ASYNC_LDS
  stageAsync(0, 0);
  async_wait0();
#else
  loadKV(0); storeKV(0);
#endif
  __syncthreads();

  for (int j = 0; j < nkt; ++j) {
    const int cur = j & 1;
    const bool more = (j + 1) < nkt;
#if ASYNC_LDS
    if (more) stageAsync(j + 1, 1 - cur);     // DMA overlaps compute below
#else
    if (more) loadKV(j + 1);
#endif

    // ---- scores S = Q * K^T (two 16-key sub-tiles) ----
    v8f s0 = vzero, s1 = vzero;
    #pragma unroll
    for (int c = 0; c < 4; ++c) {
      const _Float16* p0 = &Ks[cur][lh][c * 32 + l16 * 8];
      const _Float16* p1 = &Ks[cur][16 + lh][c * 32 + l16 * 8];
      V16 t0, t1;
      t0.h[0] = *(const v8h*)(p0); t0.h[1] = *(const v8h*)(p0 + 16);
      t1.h[0] = *(const v8h*)(p1); t1.h[1] = *(const v8h*)(p1 + 16);
      s0 = wmma_f16(qf[c], t0.v, s0);
      s1 = wmma_f16(qf[c], t1.v, s1);
    }

    // ---- causal mask + online softmax (fp32) ----
    #pragma unroll
    for (int v = 0; v < 8; ++v) {
      const int q = qrow0 + v + 8 * l16;
      float a = s0[v] * scale; if (j * 32 + lh > q)       a = -1e30f;
      float c2 = s1[v] * scale; if (j * 32 + 16 + lh > q) c2 = -1e30f;
      float tm = fmaxf(a, c2);
      tm = fmaxf(tm, __shfl_xor(tm, 1, 16));
      tm = fmaxf(tm, __shfl_xor(tm, 2, 16));
      tm = fmaxf(tm, __shfl_xor(tm, 4, 16));
      tm = fmaxf(tm, __shfl_xor(tm, 8, 16));
      float nm = fmaxf(m8[v], tm);
      float p0 = exp2f((a - nm) * LOG2E);
      float p1 = exp2f((c2 - nm) * LOG2E);
      float corr = exp2f((m8[v] - nm) * LOG2E);
      m8[v] = nm;
      float rs = p0 + p1;
      rs += __shfl_xor(rs, 1, 16);
      rs += __shfl_xor(rs, 2, 16);
      rs += __shfl_xor(rs, 4, 16);
      rs += __shfl_xor(rs, 8, 16);
      l8[v] = l8[v] * corr + rs;
      corr8[v] = corr;
      Ps[w][v + 8 * l16][lh]      = (_Float16)p0;   // P row-major [qrow][key]
      Ps[w][v + 8 * l16][16 + lh] = (_Float16)p1;
    }

    // ---- O = O*corr + P*V ----
    V16 pa;
    {
      const _Float16* pp = &Ps[w][lh][l16 ? 8 : 0];
      pa.h[0] = *(const v8h*)(pp); pa.h[1] = *(const v8h*)(pp + 16);
    }
    #pragma unroll
    for (int n = 0; n < 8; ++n) {
      v8f t = o[n];
      #pragma unroll
      for (int v = 0; v < 8; ++v) t[v] *= corr8[v];
      const _Float16* vp = &Vs[cur][n * 16 + lh][l16 * 8];
      V16 bf; bf.h[0] = *(const v8h*)(vp); bf.h[1] = *(const v8h*)(vp + 16);
      o[n] = wmma_f16(pa.v, bf.v, t);
    }

#if !ASYNC_LDS
    if (more) storeKV(1 - cur);
#endif
    async_wait0();
    __syncthreads();
  }

  // ---- normalize and store f16 attn output (token-major for final GEMM) ----
  #pragma unroll
  for (int n = 0; n < 8; ++n)
    #pragma unroll
    for (int v = 0; v < 8; ++v) {
      int srow = qrow0 + v + 8 * l16;
      size_t token = (size_t)b * S_ + srow;
      Ah[token * (H_ * D_) + h * D_ + n * 16 + lh] = (_Float16)(o[n][v] / l8[v]);
    }
}

// ---------------------------------------------------------------------------
extern "C" void kernel_launch(void* const* d_in, const int* in_sizes, int n_in,
                              void* d_out, int out_size, void* d_ws, size_t ws_size,
                              hipStream_t stream) {
  (void)in_sizes; (void)n_in; (void)out_size; (void)ws_size;
  const float* hidden = (const float*)d_in[0];
  // d_in[1] attention_mask is exactly causal -> implemented directly
  const float* cosT = (const float*)d_in[2];
  const float* sinT = (const float*)d_in[3];
  const float* Wq = (const float*)d_in[4];
  const float* Wk = (const float*)d_in[5];
  const float* Wv = (const float*)d_in[6];
  const float* Wo = (const float*)d_in[7];
  const float* qn = (const float*)d_in[8];
  const float* kn = (const float*)d_in[9];

  _Float16* Qh = (_Float16*)d_ws;                        // (B,H,S,D)   f16
  _Float16* Kh = Qh + (size_t)B_ * H_ * S_ * D_;         // (B,KVH,S,D) f16
  _Float16* Vh = Kh + (size_t)B_ * KVH_ * S_ * D_;       // (B,KVH,D,S) f16 (transposed)
  _Float16* Ah = Vh + (size_t)B_ * KVH_ * S_ * D_;       // (B*S, H*D)  f16

  dim3 blk(256);
  // Q/K/V projections (fused RMSNorm+RoPE epilogue for Q,K; V transposed)
  gemm_kernel<false, EPI_QK><<<dim3(16, 32), blk, 0, stream>>>(
      hidden, Wq, Qh, B_ * S_, H_ * D_, HID_, qn, cosT, sinT, H_);
  gemm_kernel<false, EPI_QK><<<dim3(4, 32), blk, 0, stream>>>(
      hidden, Wk, Kh, B_ * S_, KVH_ * D_, HID_, kn, cosT, sinT, KVH_);
  gemm_kernel<false, EPI_V><<<dim3(4, 32), blk, 0, stream>>>(
      hidden, Wv, Vh, B_ * S_, KVH_ * D_, HID_, nullptr, nullptr, nullptr, KVH_);
  // causal GQA flash attention
  attn_kernel<<<dim3(16, 32), blk, 0, stream>>>(Qh, Kh, Vh, Ah);
  // output projection -> fp32 out
  gemm_kernel<true, EPI_OUT><<<dim3(16, 32), blk, 0, stream>>>(
      Ah, Wo, d_out, B_ * S_, HID_, H_ * D_, nullptr, nullptr, nullptr, 0);
}
